// MultiHeadAttention_32830730010919
// MI455X (gfx1250) — matrix operand order repaired and checked by reading, not compiled, not run on hardware
//
#include <hip/hip_runtime.h>
#include <stdint.h>

// Problem dims (fixed by reference)
#define B_  2
#define S_  2048
#define E_  1024
#define H_  16
#define DK_ 64
#define M_  (B_*S_)   // 4096 rows of activations

typedef __attribute__((ext_vector_type(16))) __bf16 v16bf;
typedef __attribute__((ext_vector_type(8)))  float  v8f;
typedef __attribute__((ext_vector_type(4)))  int    v4i_;

union Frag16 { v16bf v; uint4 u[2]; };

__device__ __forceinline__ v8f wmma_bf16(v16bf a, v16bf b, v8f c) {
  // (neg_a, A, neg_b, B, c_mod, C, reuse_a, reuse_b)
  return __builtin_amdgcn_wmma_f32_16x16x32_bf16(false, a, false, b, (short)0, c, false, false);
}

#define NEG_INF (-__builtin_inff())

// ---------------------------------------------------------------------------
// Async direct-to-LDS staging (gfx1250 GLOBAL_LOAD_ASYNC_TO_LDS_B128, ASYNCcnt)
// Builtins exist on this toolchain (probe-confirmed); parameters are typed
// as pointers to 4 x i32 vectors per the compiler diagnostic.
// ---------------------------------------------------------------------------
#if defined(__has_builtin)
#  if __has_builtin(__builtin_amdgcn_global_load_async_to_lds_b128) && \
      __has_builtin(__builtin_amdgcn_s_wait_asynccnt)
#    define HAS_ASYNC_LDS 1
#  endif
#endif
#ifndef HAS_ASYNC_LDS
#  define HAS_ASYNC_LDS 0
#endif

#if HAS_ASYNC_LDS
#  define ASYNC_STAGE16(gp, lp)                                        \
     __builtin_amdgcn_global_load_async_to_lds_b128(                   \
         (v4i_*)(void*)(gp), (v4i_*)(void*)(lp), 0, 0)
#  define ASYNC_WAIT() __builtin_amdgcn_s_wait_asynccnt(0)
#else
#  define ASYNC_STAGE16(gp, lp) (*(uint4*)(lp) = *(const uint4*)(gp))
#  define ASYNC_WAIT() ((void)0)
#endif

// ---------------------------------------------------------------------------
// Kernel 1: fp32 -> bf16 conversion (activations + weights), 4 elems/thread
// ---------------------------------------------------------------------------
__global__ __launch_bounds__(256)
void cvt_f32_to_bf16(const float* __restrict__ src, __bf16* __restrict__ dst, int n) {
  int i = (blockIdx.x * 256 + threadIdx.x) * 4;
  if (i + 3 < n) {
    float4 f = *(const float4*)(src + i);
    dst[i + 0] = (__bf16)f.x;
    dst[i + 1] = (__bf16)f.y;
    dst[i + 2] = (__bf16)f.z;
    dst[i + 3] = (__bf16)f.w;
  }
}

// ---------------------------------------------------------------------------
// Kernel 2: GEMM  out[m,n] = sum_k A[m,k]*W[n,k] + bias[n]   (X @ W^T + b)
//   A: [Mdim,Kdim] bf16 row-major, W: [Ndim,Kdim] bf16 row-major
//   Block: 256 threads = 8 waves; per-WG tile 256(M) x 64(N); wave = 32 M-rows.
//   W tile double-buffered in LDS (async staged), K-step 32.
//   mode 0: store bf16 scaled into Qh/Kh layout [B,H,S,DK]
//   mode 1: store bf16 into Vt layout [B,H,DK,S]
//   mode 2: store f32 into [Mdim,Ndim] (final output, nontemporal)
// ---------------------------------------------------------------------------
__global__ __launch_bounds__(256)
void gemm_bf16_wmma(const __bf16* __restrict__ A,
                    const __bf16* __restrict__ W,
                    const float*  __restrict__ bias,
                    __bf16* __restrict__ outb,
                    float*  __restrict__ outf,
                    int Mdim, int Ndim, int Kdim,
                    float scale, int mode)
{
  __shared__ uint4 tileW_raw[2][64 * 40 * 2 / 16];   // 2 x (64 rows x 40 halves)
  const int tid   = threadIdx.x;
  const int wave  = tid >> 5;
  const int lane  = tid & 31;
  const int lhalf = lane & 15;
  const int hi    = lane >> 4;                       // 0: lanes 0-15, 1: lanes 16-31
  const int m0    = blockIdx.x * 256 + wave * 32;
  const int n0    = blockIdx.y * 64;

  // cooperative W-tile staging coordinates (256 thr cover 64 rows x 32 halves)
  const int wr = tid >> 2;
  const int wc = (tid & 3) * 8;

  const __bf16* __restrict__ Arow0 = A + (size_t)(m0 + lhalf) * Kdim;
  const __bf16* __restrict__ Arow1 = A + (size_t)(m0 + 16 + lhalf) * Kdim;

  v8f acc[2][4] = {};

  // prologue: stage first W tile into buffer 0
  ASYNC_STAGE16(W + (size_t)(n0 + wr) * Kdim + wc, &((__bf16*)tileW_raw[0])[wr * 40 + wc]);
  ASYNC_WAIT();
  __syncthreads();

  const int nIter = Kdim >> 5;
  for (int it = 0; it < nIter; ++it) {
    const int k0  = it << 5;
    const int cur = it & 1;
    __bf16* tileW = (__bf16*)tileW_raw[cur];

    // stage next tile into the other buffer while we compute on this one
    if (it + 1 < nIter) {
      ASYNC_STAGE16(W + (size_t)(n0 + wr) * Kdim + (k0 + 32) + wc,
                    &((__bf16*)tileW_raw[cur ^ 1])[wr * 40 + wc]);
      __builtin_prefetch(Arow0 + k0 + 32, 0, 1);
    }

    // A fragments: 16-bit A layout (interleaved 8-half chunks per lane half)
    Frag16 fa0, fa1;
    fa0.u[0] = *(const uint4*)(Arow0 + k0 + hi * 8);
    fa0.u[1] = *(const uint4*)(Arow0 + k0 + hi * 8 + 16);
    fa1.u[0] = *(const uint4*)(Arow1 + k0 + hi * 8);
    fa1.u[1] = *(const uint4*)(Arow1 + k0 + hi * 8 + 16);

#pragma unroll
    for (int f = 0; f < 4; ++f) {
      // B fragment: lane n holds interleaved 8-half K chunks of column n
      const __bf16* wrow = &tileW[(f * 16 + lhalf) * 40 + hi * 8];
      Frag16 fb;
      fb.u[0] = *(const uint4*)(wrow);
      fb.u[1] = *(const uint4*)(wrow + 16);
      acc[0][f] = wmma_bf16(fa0.v, fb.v, acc[0][f]);
      acc[1][f] = wmma_bf16(fa1.v, fb.v, acc[1][f]);
    }

    ASYNC_WAIT();
    __syncthreads();
  }

  // Epilogue: bias add + store. C/D layout: lane%16 = n, vgpr(+8 for hi) = m.
#pragma unroll
  for (int g = 0; g < 2; ++g) {
#pragma unroll
    for (int f = 0; f < 4; ++f) {
      const int n = n0 + f * 16 + lhalf;
      const float bv = bias[n];
#pragma unroll
      for (int r = 0; r < 8; ++r) {
        const int m = m0 + g * 16 + r + hi * 8;
        const float v = acc[g][f][r] + bv;
        if (mode == 2) {
          __builtin_nontemporal_store(v, &outf[(size_t)m * Ndim + n]);
        } else {
          const __bf16 o = (__bf16)(v * scale);
          const int b = m >> 11, s = m & 2047;
          const int h = n >> 6,  dk = n & 63;
          size_t off;
          if (mode == 0) off = ((((size_t)b * H_ + h) * S_ + s) * DK_ + dk);  // [B,H,S,DK]
          else           off = ((((size_t)b * H_ + h) * DK_ + dk) * S_ + s);  // [B,H,DK,S]
          outb[off] = o;
        }
      }
    }
  }
}

// ---------------------------------------------------------------------------
// Kernel 3: flash attention per (b,h). 256 thr = 8 waves; wave owns 16 q rows.
//   Qh pre-scaled by log2(e)/sqrt(DK) so softmax runs in exp2 domain.
//   Scores computed transposed (S^T = K * Q^T): M=keys, N=queries, so the
//   f32 D-frag per-lane data is exactly the P A-fragment for O = P*V.
//   K/V blocks double-buffered in LDS, async staged.
// ---------------------------------------------------------------------------
__global__ __launch_bounds__(256)
void attention_wmma(const __bf16* __restrict__ Qh,   // [B,H,S,DK] (prescaled)
                    const __bf16* __restrict__ Kh,   // [B,H,S,DK]
                    const __bf16* __restrict__ Vt,   // [B,H,DK,S]
                    const int*    __restrict__ mask, // [B,S,S], 0 => masked
                    __bf16* __restrict__ X)          // [B,S,E]
{
  __shared__ uint4 ldsK_raw[2][32 * 72 * 2 / 16];    // 32 keys x 72 halves (pad 64->72)
  __shared__ uint4 ldsV_raw[2][64 * 40 * 2 / 16];    // 64 d    x 40 halves (pad 32->40)

  const int tid   = threadIdx.x;
  const int wave  = tid >> 5;
  const int lane  = tid & 31;
  const int lhalf = lane & 15;
  const int hi    = lane >> 4;
  const int b = blockIdx.z, h = blockIdx.y;
  const int q0 = blockIdx.x * 128 + wave * 16;
  const int q  = q0 + lhalf;

  const __bf16* __restrict__ Qrow  = Qh + (((size_t)b * H_ + h) * S_ + q) * DK_;
  const __bf16* __restrict__ Kbase = Kh + (((size_t)b * H_ + h) * S_) * DK_;
  const __bf16* __restrict__ Vbase = Vt + (((size_t)b * H_ + h) * DK_) * S_;
  const int*    __restrict__ mrow  = mask + ((size_t)b * S_ + q) * S_;

  // cooperative staging coordinates
  const int kr = tid >> 3, kc = (tid & 7) * 8;       // K block: 32 rows x 64 halves
  const int vr = tid >> 2, vc = (tid & 3) * 8;       // V block: 64 rows x 32 halves

  // Q as B-matrix fragments (two halves of d), interleaved 8-half chunk loads
  Frag16 bq0, bq1;
  bq0.u[0] = *(const uint4*)(Qrow + hi * 8);
  bq0.u[1] = *(const uint4*)(Qrow + hi * 8 + 16);
  bq1.u[0] = *(const uint4*)(Qrow + 32 + hi * 8);
  bq1.u[1] = *(const uint4*)(Qrow + 32 + hi * 8 + 16);

  v8f   acc[4] = {};
  float mrun = NEG_INF;
  float lrun = 0.f;

  // prologue: stage first K/V block into buffer 0
  ASYNC_STAGE16(Kbase + (size_t)kr * DK_ + kc, &((__bf16*)ldsK_raw[0])[kr * 72 + kc]);
  ASYNC_STAGE16(Vbase + (size_t)vr * S_ + vc,  &((__bf16*)ldsV_raw[0])[vr * 40 + vc]);
  ASYNC_WAIT();
  __syncthreads();

  const int nIter = S_ >> 5;
  for (int it = 0; it < nIter; ++it) {
    const int kk0 = it << 5;
    const int cur = it & 1;
    __bf16* ldsK = (__bf16*)ldsK_raw[cur];
    __bf16* ldsV = (__bf16*)ldsV_raw[cur];

    // stage next K/V block while computing on the current one
    if (it + 1 < nIter) {
      ASYNC_STAGE16(Kbase + (size_t)(kk0 + 32 + kr) * DK_ + kc,
                    &((__bf16*)ldsK_raw[cur ^ 1])[kr * 72 + kc]);
      ASYNC_STAGE16(Vbase + (size_t)vr * S_ + (kk0 + 32) + vc,
                    &((__bf16*)ldsV_raw[cur ^ 1])[vr * 40 + vc]);
    }

    // ---- scores (transposed): St[kh] = K(16 keys) x Q^T, kh = key half
    v8f st[2];
#pragma unroll
    for (int kh = 0; kh < 2; ++kh) {
      const __bf16* krow = &ldsK[(kh * 16 + lhalf) * 72];
      Frag16 a0, a1;
      a0.u[0] = *(const uint4*)(krow + hi * 8);
      a0.u[1] = *(const uint4*)(krow + hi * 8 + 16);
      a1.u[0] = *(const uint4*)(krow + 32 + hi * 8);
      a1.u[1] = *(const uint4*)(krow + 32 + hi * 8 + 16);
      v8f c = {};
      c = wmma_bf16(a0.v, bq0.v, c);
      c = wmma_bf16(a1.v, bq1.v, c);
      st[kh] = c;
    }

    // ---- key padding mask: st[kh][r] <-> key kk0 + kh*16 + hi*8 + r
#pragma unroll
    for (int kh = 0; kh < 2; ++kh) {
      const int kb = kk0 + kh * 16 + hi * 8;
      int4 mA = *(const int4*)(mrow + kb);
      int4 mB = *(const int4*)(mrow + kb + 4);
      if (mA.x == 0) st[kh][0] = NEG_INF;
      if (mA.y == 0) st[kh][1] = NEG_INF;
      if (mA.z == 0) st[kh][2] = NEG_INF;
      if (mA.w == 0) st[kh][3] = NEG_INF;
      if (mB.x == 0) st[kh][4] = NEG_INF;
      if (mB.y == 0) st[kh][5] = NEG_INF;
      if (mB.z == 0) st[kh][6] = NEG_INF;
      if (mB.w == 0) st[kh][7] = NEG_INF;
    }

    // ---- online softmax (exp2 domain; log2e folded into Q)
    float bm = NEG_INF;
#pragma unroll
    for (int r = 0; r < 8; ++r) { bm = fmaxf(bm, st[0][r]); bm = fmaxf(bm, st[1][r]); }
    bm = fmaxf(bm, __shfl_xor(bm, 16, 32));
    const float mnew  = fmaxf(mrun, bm);
    const float alpha = __builtin_amdgcn_exp2f(mrun - mnew);

    Frag16 pa;                                   // P as A-fragment, packed in-lane
    float bl = 0.f;
#pragma unroll
    for (int r = 0; r < 8; ++r) {
      float p0 = __builtin_amdgcn_exp2f(st[0][r] - mnew);
      float p1 = __builtin_amdgcn_exp2f(st[1][r] - mnew);
      bl += p0 + p1;
      pa.v[r]     = (__bf16)p0;
      pa.v[r + 8] = (__bf16)p1;
    }
    bl += __shfl_xor(bl, 16, 32);
    lrun = lrun * alpha + bl;
    mrun = mnew;

    // ---- rescale accumulators: row index q0 + r + hi*8 lives in lane r + hi*8
#pragma unroll
    for (int r = 0; r < 8; ++r) {
      const float ar = __shfl(alpha, r + hi * 8, 32);
#pragma unroll
      for (int f = 0; f < 4; ++f) acc[f][r] *= ar;
    }

    // ---- O += P * V  (B frag = interleaved 8-half Vt row chunks from LDS)
#pragma unroll
    for (int f = 0; f < 4; ++f) {
      const __bf16* vrow = &ldsV[(f * 16 + lhalf) * 40 + hi * 8];
      Frag16 vb;
      vb.u[0] = *(const uint4*)(vrow);
      vb.u[1] = *(const uint4*)(vrow + 16);
      acc[f] = wmma_bf16(pa.v, vb.v, acc[f]);
    }

    ASYNC_WAIT();
    __syncthreads();
  }

  // ---- finalize: divide by row sum, store bf16 into X[b, q, h*64 + d]
  const float linv = __builtin_amdgcn_rcpf(lrun);
#pragma unroll
  for (int r = 0; r < 8; ++r) {
    const float lr = __shfl(linv, r + hi * 8, 32);
    const int   qq = q0 + r + hi * 8;
#pragma unroll
    for (int f = 0; f < 4; ++f) {
      X[((size_t)b * S_ + qq) * E_ + h * DK_ + f * 16 + lhalf] = (__bf16)(acc[f][r] * lr);
    }
  }
}

// ---------------------------------------------------------------------------
// Host launcher
// ---------------------------------------------------------------------------
extern "C" void kernel_launch(void* const* d_in, const int* in_sizes, int n_in,
                              void* d_out, int out_size, void* d_ws, size_t ws_size,
                              hipStream_t stream) {
  const float* qin = (const float*)d_in[0];
  const float* kin = (const float*)d_in[1];
  const float* vin = (const float*)d_in[2];
  const int*   msk = (const int*)  d_in[3];
  const float* Wq  = (const float*)d_in[4];
  const float* bq  = (const float*)d_in[5];
  const float* Wk  = (const float*)d_in[6];
  const float* bk  = (const float*)d_in[7];
  const float* Wv  = (const float*)d_in[8];
  const float* bv  = (const float*)d_in[9];
  const float* Wo  = (const float*)d_in[10];
  const float* bo  = (const float*)d_in[11];

  const size_t ACT = (size_t)M_ * E_;   // 4,194,304 elems
  const size_t WEL = (size_t)E_ * E_;   // 1,048,576 elems

  __bf16* ws  = (__bf16*)d_ws;
  __bf16* Xq  = ws;
  __bf16* Xk  = Xq  + ACT;
  __bf16* Xv  = Xk  + ACT;
  __bf16* Wqb = Xv  + ACT;
  __bf16* Wkb = Wqb + WEL;
  __bf16* Wvb = Wkb + WEL;
  __bf16* Wob = Wvb + WEL;
  __bf16* Qh  = Wob + WEL;
  __bf16* Kh  = Qh  + ACT;
  __bf16* Vt  = Kh  + ACT;
  __bf16* Xa  = Vt  + ACT;   // total 64 MiB of bf16 workspace

  // fp32 -> bf16
  cvt_f32_to_bf16<<<dim3(ACT / 1024), 256, 0, stream>>>(qin, Xq, (int)ACT);
  cvt_f32_to_bf16<<<dim3(ACT / 1024), 256, 0, stream>>>(kin, Xk, (int)ACT);
  cvt_f32_to_bf16<<<dim3(ACT / 1024), 256, 0, stream>>>(vin, Xv, (int)ACT);
  cvt_f32_to_bf16<<<dim3(WEL / 1024), 256, 0, stream>>>(Wq, Wqb, (int)WEL);
  cvt_f32_to_bf16<<<dim3(WEL / 1024), 256, 0, stream>>>(Wk, Wkb, (int)WEL);
  cvt_f32_to_bf16<<<dim3(WEL / 1024), 256, 0, stream>>>(Wv, Wvb, (int)WEL);
  cvt_f32_to_bf16<<<dim3(WEL / 1024), 256, 0, stream>>>(Wo, Wob, (int)WEL);

  const dim3 gg(M_ / 256, E_ / 64);
  // fold softmax scale (1/8, exact power of 2) and log2(e) into Q
  const float QSCALE = 1.4426950408889634f * 0.125f;
  gemm_bf16_wmma<<<gg, 256, 0, stream>>>(Xq, Wqb, bq, Qh, nullptr, M_, E_, E_, QSCALE, 0);
  gemm_bf16_wmma<<<gg, 256, 0, stream>>>(Xk, Wkb, bk, Kh, nullptr, M_, E_, E_, 1.0f,   0);
  gemm_bf16_wmma<<<gg, 256, 0, stream>>>(Xv, Wvb, bv, Vt, nullptr, M_, E_, E_, 1.0f,   1);

  attention_wmma<<<dim3(S_ / 128, H_, B_), 256, 0, stream>>>(Qh, Kh, Vt, msk, Xa);

  gemm_bf16_wmma<<<gg, 256, 0, stream>>>(Xa, Wob, bo, nullptr, (float*)d_out, M_, E_, E_, 1.0f, 2);
}
